// PatchMerging3D_13675175870627
// MI455X (gfx1250) — matrix operand order repaired and checked by reading, not compiled, not run on hardware
//
#include <hip/hip_runtime.h>
#include <math.h>

// ---------------- problem constants (from reference) ----------------
constexpr int B_    = 4;
constexpr int NPER  = 60000;
constexpr int C_    = 128;
constexpr int D_    = 32;
constexpr int H_    = 400;
constexpr int W_    = 352;
constexpr int KTOP  = 12000;                 // NPER * 0.2
constexpr int N_    = B_ * NPER;             // 240000
constexpr int NTOT  = N_ + B_ * 3 * KTOP;    // 384000
constexpr int GRID_SZ = B_ * D_ * H_ * W_;   // 18,022,400
constexpr int TK_LEN = 65536;                // pow2 >= NPER
constexpr int MG_LEN = 524288;               // pow2 >= NTOT
// down-scale merge constants
constexpr int SZ_   = 16;                    // ceil(D/2)
constexpr int SYZ_  = SZ_ * (H_ / 2);        // 3200
constexpr int SXYZ_ = SYZ_ * (W_ / 2);       // 563200

typedef __attribute__((ext_vector_type(16))) __bf16 v16bf;
typedef __attribute__((ext_vector_type(8)))  float  v8f;

union AFrag { v16bf v; unsigned int u[8]; };

__device__ __forceinline__ unsigned short f32_to_bf16(float f) {
    unsigned int u = __float_as_uint(f);
    unsigned int r = 0x7FFFu + ((u >> 16) & 1u);
    return (unsigned short)((u + r) >> 16);
}

// ---------------- prep kernels ----------------
__global__ void k_fill_grid(int* g, int n) {
    int i = blockIdx.x * blockDim.x + threadIdx.x;
    if (i < n) g[i] = -1;
}

__global__ void k_scatter(const int* __restrict__ idx, int* __restrict__ g, int n) {
    int i = blockIdx.x * blockDim.x + threadIdx.x;
    if (i >= n) return;
    int b = idx[i * 4 + 0], z = idx[i * 4 + 1], y = idx[i * 4 + 2], x = idx[i * 4 + 3];
    g[((b * D_ + z) * H_ + y) * W_ + x] = i;
}

__global__ void k_cvt_feat(const float* __restrict__ f, unsigned short* __restrict__ fb, int n) {
    int i = blockIdx.x * blockDim.x + threadIdx.x;
    if (i < n) fb[i] = f32_to_bf16(f[i]);
}

// Wt[k][n][kk] = W[k][kk][n]  (bf16, transposed so B fragments are contiguous in K)
__global__ void k_cvt_w(const float* __restrict__ W, unsigned short* __restrict__ Wt, int n) {
    int t = blockIdx.x * blockDim.x + threadIdx.x;
    if (t >= n) return;
    int k = t / (C_ * C_);
    int r = t - k * C_ * C_;
    int nn = r / C_;
    int kk = r - nn * C_;
    Wt[t] = f32_to_bf16(W[(k * C_ + kk) * C_ + nn]);
}

// ---------------- fused subm-conv (WMMA bf16) + LayerNorm + GELU + att ----------------
// 32-row M-tile per block: each wave owns one 16-col slice and two 16x16 accumulators,
// so every B fragment loaded from L2 feeds two WMMAs (halves weight re-read traffic).
__global__ void __launch_bounds__(256)
k_conv(const int* __restrict__ idx, const int* __restrict__ grid,
       const unsigned int* __restrict__ fbu,        // bf16 features packed as uint (2/elem)
       const unsigned short* __restrict__ wt,       // bf16 Wt [27][128][128]
       const float* __restrict__ gamma, const float* __restrict__ beta,
       float* __restrict__ xout, float* __restrict__ att) {
    __shared__ __align__(32) unsigned short sA[32][144];  // 32 gathered rows, bf16
    __shared__ float sX[32][132];
    __shared__ int   sNid[32];
    __shared__ float sMu[32], sRs[32];

    const int row0 = blockIdx.x * 32;
    const int tid  = threadIdx.x;
    const int wv   = tid >> 5;        // 0..7 : output-column tile
    const int ln   = tid & 31;
    const int m    = ln & 15;
    const int half = ln >> 4;         // 0/1

    v8f acc0 = {};
    v8f acc1 = {};

    for (int kk = 0; kk < 27; ++kk) {
        const int dz = kk / 9 - 1, dy = (kk / 3) % 3 - 1, dx = kk % 3 - 1;
        if (tid < 32) {
            const int r = row0 + tid;
            const int b = idx[r * 4 + 0];
            const int z = idx[r * 4 + 1] + dz;
            const int y = idx[r * 4 + 2] + dy;
            const int x = idx[r * 4 + 3] + dx;
            int nid = -1;
            if (z >= 0 && z < D_ && y >= 0 && y < H_ && x >= 0 && x < W_)
                nid = grid[((b * D_ + z) * H_ + y) * W_ + x];
            sNid[tid] = nid;
        }
        __syncthreads();
        // gather 32 rows x 128 bf16 (64 uints/row) into LDS
        for (int e = tid; e < 32 * 64; e += 256) {
            const int r  = e >> 6;
            const int cu = e & 63;
            const int nid = sNid[r];
            unsigned int val = 0u;
            if (nid >= 0) val = fbu[(size_t)nid * 64 + cu];
            *(unsigned int*)&sA[r][cu * 2] = val;
        }
        __syncthreads();
        // D(16x16) += A(16x32) * B(32x16), 4 K-steps of 32, two M-tiles share B
        #pragma unroll
        for (int ks = 0; ks < 4; ++ks) {
            // B: lane holds 16 contiguous K for column n
            const int n = (wv << 4) + m;
            const v16bf b = *(const v16bf*)(wt + (((size_t)kk * 128 + n) * 128 + ks * 32 + half * 16));
            // A: lane(m,half) holds 16 contiguous K [16*half, 16*half+16)
            // of this 32-wide K step (matches B).
            AFrag a0, a1;
            const unsigned int* p0 = (const unsigned int*)&sA[m][ks * 32 + half * 16];
            const unsigned int* p1 = (const unsigned int*)&sA[m][ks * 32 + half * 16 + 8];
            const unsigned int* q0 = (const unsigned int*)&sA[16 + m][ks * 32 + half * 16];
            const unsigned int* q1 = (const unsigned int*)&sA[16 + m][ks * 32 + half * 16 + 8];
            #pragma unroll
            for (int q = 0; q < 4; ++q) {
                a0.u[q] = p0[q]; a0.u[4 + q] = p1[q];
                a1.u[q] = q0[q]; a1.u[4 + q] = q1[q];
            }
            acc0 = __builtin_amdgcn_wmma_f32_16x16x32_bf16(false, a0.v, false, b,
                                                           (short)0, acc0, false, false);
            acc1 = __builtin_amdgcn_wmma_f32_16x16x32_bf16(false, a1.v, false, b,
                                                           (short)0, acc1, false, false);
        }
        __syncthreads();
    }

    // spill accumulators to LDS f32 tile (C/D layout: VGPR v -> M=v+8*half, N=m)
    #pragma unroll
    for (int v = 0; v < 8; ++v) {
        sX[v + half * 8][(wv << 4) + m]      = acc0[v];
        sX[16 + v + half * 8][(wv << 4) + m] = acc1[v];
    }
    __syncthreads();

    // LayerNorm stats per row
    if (tid < 32) {
        float s = 0.f, sq = 0.f;
        for (int c = 0; c < 128; ++c) { float t = sX[tid][c]; s += t; sq += t * t; }
        const float mu = s * (1.0f / 128.0f);
        sMu[tid] = mu;
        sRs[tid] = rsqrtf(sq * (1.0f / 128.0f) - mu * mu + 1e-5f);
    }
    __syncthreads();
    // normalize + affine + exact GELU, write x
    for (int e = tid; e < 4096; e += 256) {
        const int r = e >> 7, c = e & 127;
        float v = (sX[r][c] - sMu[r]) * sRs[r] * gamma[c] + beta[c];
        v = 0.5f * v * (1.0f + erff(v * 0.70710678118654752f));
        sX[r][c] = v;
        xout[(size_t)(row0 + r) * 128 + c] = v;
    }
    __syncthreads();
    if (tid < 32) {
        float s = 0.f;
        for (int c = 0; c < 128; ++c) s += sX[tid][c];
        att[row0 + tid] = s * (1.0f / 128.0f);
    }
}

// ---------------- top-k via batched bitonic sort ----------------
__global__ void k_tkeys(const float* __restrict__ att, unsigned long long* __restrict__ keys) {
    int t = blockIdx.x * blockDim.x + threadIdx.x;
    if (t >= B_ * TK_LEN) return;
    const int b = t / TK_LEN, i = t - b * TK_LEN;
    unsigned long long key = 0ull;
    if (i < NPER) {
        unsigned int u = __float_as_uint(att[b * NPER + i]);
        u = (u & 0x80000000u) ? ~u : (u | 0x80000000u);   // order-preserving map
        key = ((unsigned long long)u << 32) | (unsigned long long)(0xFFFFFFFFu - (unsigned int)i);
    }
    keys[t] = key;
}

__global__ void k_bitonic(unsigned long long* __restrict__ keys, int len, int j, int k, int nb, int asc) {
    int t = blockIdx.x * blockDim.x + threadIdx.x;
    if (t >= nb * len) return;
    const int b = t / len, i = t - b * len;
    const int ixj = i ^ j;
    if (ixj <= i) return;
    unsigned long long* base = keys + (size_t)b * len;
    const unsigned long long a = base[i], c = base[ixj];
    const bool up  = ((i & k) == 0);
    const bool dir = (up == (asc != 0));
    if ((a > c) == dir) { base[i] = c; base[ixj] = a; }
}

__global__ void k_copy_coords(const int* __restrict__ idx, int* __restrict__ coords, int n4) {
    int i = blockIdx.x * blockDim.x + threadIdx.x;
    if (i < n4) coords[i] = idx[i];
}

__global__ void k_diffuse(const unsigned long long* __restrict__ tkeys,
                          const int* __restrict__ idx, int* __restrict__ coords) {
    int t = blockIdx.x * blockDim.x + threadIdx.x;
    if (t >= B_ * KTOP) return;
    const int b = t / KTOP, q = t - b * KTOP;
    const unsigned long long key = tkeys[(size_t)b * TK_LEN + q];
    const int row = (int)(0xFFFFFFFFu - (unsigned int)(key & 0xFFFFFFFFull));
    const int* s = idx + ((size_t)b * NPER + row) * 4;
    const int d = s[1], h = s[2], w = s[3];
    const int dh = (h > 200) ? 1 : -1;   // sign(h - 200.5)
    const int dw = (w > 176) ? 1 : -1;   // sign(w - 176.5)
    const int dc = min(max(d, 0), D_ - 1);
    const int hc = min(max(h, 0), H_ - 1);
    const int wc = min(max(w, 0), W_ - 1);
    const int hs = min(max(h + dh, 0), H_ - 1);
    const int ws = min(max(w + dw, 0), W_ - 1);
    const int base = N_ + b * 3 * KTOP;
    int* o0 = coords + (size_t)(base + q) * 4;
    o0[0] = b; o0[1] = dc; o0[2] = hc; o0[3] = ws;
    int* o1 = coords + (size_t)(base + KTOP + q) * 4;
    o1[0] = b; o1[1] = dc; o1[2] = hs; o1[3] = wc;
    int* o2 = coords + (size_t)(base + 2 * KTOP + q) * 4;
    o2[0] = b; o2[1] = dc; o2[2] = hs; o2[3] = ws;
}

// ---------------- merge (unique + segment sum) ----------------
__global__ void k_mkeys(const int* __restrict__ coords, unsigned long long* __restrict__ keys) {
    int i = blockIdx.x * blockDim.x + threadIdx.x;
    if (i >= MG_LEN) return;
    unsigned long long key = ~0ull;
    if (i < NTOT) {
        const int b = coords[i * 4 + 0];
        const int cd = coords[i * 4 + 1] >> 1;
        const int ch = coords[i * 4 + 2] >> 1;
        const int cw = coords[i * 4 + 3] >> 1;
        const int merge = b * SXYZ_ + cw * SYZ_ + ch * SZ_ + cd;
        key = ((unsigned long long)(unsigned int)merge << 32) | (unsigned int)i;  // stable ties
    }
    keys[i] = key;
}

__global__ void k_flags(const unsigned long long* __restrict__ S, int* __restrict__ seg) {
    int p = blockIdx.x * blockDim.x + threadIdx.x;
    if (p >= NTOT) return;
    int f = 0;
    if (p > 0) f = ((unsigned int)(S[p] >> 32) != (unsigned int)(S[p - 1] >> 32)) ? 1 : 0;
    seg[p] = f;
}

#define SCAN_BLK 1024
__global__ void k_scan1(int* __restrict__ data, int n, int* __restrict__ bsums) {
    __shared__ int sh[256];
    const int base = blockIdx.x * SCAN_BLK;
    const int tid = threadIdx.x;
    int v[4]; int s = 0;
    #pragma unroll
    for (int q = 0; q < 4; ++q) {
        const int p = base + tid * 4 + q;
        const int x = (p < n) ? data[p] : 0;
        s += x; v[q] = s;
    }
    sh[tid] = s;
    __syncthreads();
    for (int off = 1; off < 256; off <<= 1) {
        const int t2 = (tid >= off) ? sh[tid - off] : 0;
        __syncthreads();
        sh[tid] += t2;
        __syncthreads();
    }
    const int prefix = (tid > 0) ? sh[tid - 1] : 0;
    #pragma unroll
    for (int q = 0; q < 4; ++q) {
        const int p = base + tid * 4 + q;
        if (p < n) data[p] = v[q] + prefix;
    }
    if (tid == 255) bsums[blockIdx.x] = sh[255];
}

__global__ void k_scan2(int* __restrict__ bsums, int nb) {
    if (blockIdx.x == 0 && threadIdx.x == 0) {
        int acc = 0;
        for (int i = 0; i < nb; ++i) { const int t = bsums[i]; bsums[i] = acc; acc += t; }
    }
}

__global__ void k_scan3(int* __restrict__ data, int n, const int* __restrict__ bsums) {
    int p = blockIdx.x * blockDim.x + threadIdx.x;
    if (p < n) data[p] += bsums[p / SCAN_BLK];
}

__global__ void k_init_unq(int* __restrict__ unq) {
    int i = blockIdx.x * blockDim.x + threadIdx.x;
    if (i < NTOT) unq[i] = -1;
}

__global__ void k_postsort(const unsigned long long* __restrict__ S, const int* __restrict__ seg,
                           int* __restrict__ uinv, int* __restrict__ unq,
                           int* __restrict__ segB, int* __restrict__ segE, int* __restrict__ dnum) {
    int p = blockIdx.x * blockDim.x + threadIdx.x;
    if (p >= NTOT) return;
    const int s = seg[p];
    const unsigned int ord = (unsigned int)(S[p] & 0xFFFFFFFFull);
    uinv[ord] = s;
    const bool start = (p == 0) || (s > seg[p - 1]);
    if (start) {
        unq[s]  = (int)(unsigned int)(S[p] >> 32);
        segB[s] = p;
        if (p > 0) segE[s - 1] = p;
    }
    if (p == NTOT - 1) { segE[s] = NTOT; *dnum = s + 1; }
}

__global__ void __launch_bounds__(128)
k_segsum(const unsigned long long* __restrict__ S, const int* __restrict__ segB,
         const int* __restrict__ segE, const float* __restrict__ x,
         const int* __restrict__ dnum, float* __restrict__ xm) {
    const int s = blockIdx.x;
    const int c = threadIdx.x;
    if (s >= *dnum) { xm[(size_t)s * 128 + c] = 0.f; return; }
    const int beg = segB[s], end = segE[s];
    float acc = 0.f;
    for (int p = beg; p < end; ++p) {
        const unsigned int ord = (unsigned int)(S[p] & 0xFFFFFFFFull);
        if (ord < (unsigned int)N_) acc += x[(size_t)ord * 128 + c];  // diffusion rows contribute 0
    }
    xm[(size_t)s * 128 + c] = acc;
}

__global__ void k_finalize(const int* __restrict__ unq, const int* __restrict__ uinv,
                           const int* __restrict__ dnum, float* __restrict__ out) {
    int i = blockIdx.x * blockDim.x + threadIdx.x;
    if (i >= NTOT) return;
    const int u = unq[i];
    int vb = -1, vd = -1, vh = -1, vw = -1;
    if (u >= 0) {
        vb = u / SXYZ_;
        vw = (u % SXYZ_) / SYZ_;
        vh = (u % SYZ_) / SZ_;
        vd = u % SZ_;
    }
    float* vox = out + (size_t)NTOT * 128;
    vox[(size_t)i * 4 + 0] = (float)vb;
    vox[(size_t)i * 4 + 1] = (float)vd;
    vox[(size_t)i * 4 + 2] = (float)vh;
    vox[(size_t)i * 4 + 3] = (float)vw;
    out[(size_t)NTOT * 132 + i] = (float)uinv[i];
    if (i == 0) out[(size_t)NTOT * 133] = (float)(*dnum);
}

// ---------------- host orchestration ----------------
extern "C" void kernel_launch(void* const* d_in, const int* in_sizes, int n_in,
                              void* d_out, int out_size, void* d_ws, size_t ws_size,
                              hipStream_t stream) {
    (void)in_sizes; (void)n_in; (void)out_size; (void)ws_size;
    const float* features = (const float*)d_in[0];
    const int*   indices  = (const int*)d_in[1];
    const float* W        = (const float*)d_in[2];
    const float* gamma    = (const float*)d_in[3];
    const float* beta     = (const float*)d_in[4];
    float* out = (float*)d_out;

    char* ws = (char*)d_ws;
    size_t off = 0;
    auto carve = [&](size_t bytes) -> char* {
        char* p = ws + off;
        off += (bytes + 255) & ~(size_t)255;
        return p;
    };
    int*                grid   = (int*)carve((size_t)GRID_SZ * 4);
    unsigned short*     fb     = (unsigned short*)carve((size_t)N_ * C_ * 2);
    unsigned short*     wt     = (unsigned short*)carve((size_t)27 * C_ * C_ * 2);
    float*              x      = (float*)carve((size_t)N_ * C_ * 4);
    float*              att    = (float*)carve((size_t)N_ * 4);
    unsigned long long* tkeys  = (unsigned long long*)carve((size_t)B_ * TK_LEN * 8);
    int*                coords = (int*)carve((size_t)NTOT * 4 * 4);
    unsigned long long* mkeys  = (unsigned long long*)carve((size_t)MG_LEN * 8);
    int*                seg    = (int*)carve((size_t)MG_LEN * 4);
    int*                bsums  = (int*)carve((size_t)1024 * 4);
    int*                unq    = (int*)carve((size_t)NTOT * 4);
    int*                uinv   = (int*)carve((size_t)NTOT * 4);
    int*                segB   = (int*)carve((size_t)NTOT * 4);
    int*                segE   = (int*)carve((size_t)NTOT * 4);
    int*                dnum   = (int*)carve(4);

    // 1) hash grid
    k_fill_grid<<<(GRID_SZ + 255) / 256, 256, 0, stream>>>(grid, GRID_SZ);
    k_scatter<<<(N_ + 255) / 256, 256, 0, stream>>>(indices, grid, N_);
    // 2) bf16 conversions
    k_cvt_feat<<<(N_ * C_ + 255) / 256, 256, 0, stream>>>(features, fb, N_ * C_);
    k_cvt_w<<<(27 * C_ * C_ + 255) / 256, 256, 0, stream>>>(W, wt, 27 * C_ * C_);
    // 3) fused WMMA conv + LN + GELU + att (32-row M-tile)
    k_conv<<<N_ / 32, 256, 0, stream>>>(indices, grid, (const unsigned int*)fb, wt,
                                        gamma, beta, x, att);
    // 4) top-k per batch: bitonic sort descending
    k_tkeys<<<(B_ * TK_LEN + 255) / 256, 256, 0, stream>>>(att, tkeys);
    for (int k = 2; k <= TK_LEN; k <<= 1)
        for (int j = k >> 1; j > 0; j >>= 1)
            k_bitonic<<<(B_ * TK_LEN + 255) / 256, 256, 0, stream>>>(tkeys, TK_LEN, j, k, B_, 0);
    // 5) diffusion coords
    k_copy_coords<<<(N_ * 4 + 255) / 256, 256, 0, stream>>>(indices, coords, N_ * 4);
    k_diffuse<<<(B_ * KTOP + 255) / 256, 256, 0, stream>>>(tkeys, indices, coords);
    // 6) merge keys + ascending sort (stable via index in low bits)
    k_mkeys<<<MG_LEN / 256, 256, 0, stream>>>(coords, mkeys);
    for (int k = 2; k <= MG_LEN; k <<= 1)
        for (int j = k >> 1; j > 0; j >>= 1)
            k_bitonic<<<(MG_LEN + 255) / 256, 256, 0, stream>>>(mkeys, MG_LEN, j, k, 1, 1);
    // 7) segment ids: flags -> inclusive scan
    k_flags<<<(NTOT + 255) / 256, 256, 0, stream>>>(mkeys, seg);
    const int nScanBlk = (NTOT + SCAN_BLK - 1) / SCAN_BLK;
    k_scan1<<<nScanBlk, 256, 0, stream>>>(seg, NTOT, bsums);
    k_scan2<<<1, 1, 0, stream>>>(bsums, nScanBlk);
    k_scan3<<<(NTOT + 255) / 256, 256, 0, stream>>>(seg, NTOT, bsums);
    // 8) unique / inverse / segment ranges
    k_init_unq<<<(NTOT + 255) / 256, 256, 0, stream>>>(unq);
    k_postsort<<<(NTOT + 255) / 256, 256, 0, stream>>>(mkeys, seg, uinv, unq, segB, segE, dnum);
    // 9) deterministic segment sum straight into d_out (x_merge)
    k_segsum<<<NTOT, 128, 0, stream>>>(mkeys, segB, segE, x, dnum, out);
    // 10) voxel decode + unq_inv + num_unique
    k_finalize<<<(NTOT + 255) / 256, 256, 0, stream>>>(unq, uinv, dnum, out);
}
